// ScatteringAttentionLayer_87479893885738
// MI455X (gfx1250) — matrix operand order repaired and checked by reading, not compiled, not run on hardware
//
#include <hip/hip_runtime.h>
#include <hip/hip_bf16.h>

// Problem constants (from reference)
#define NN   20000
#define EE   640000
#define FIN  128
#define DD   16
#define HH   8
#define NF   (NN * FIN)           // 2,560,000 floats per node-feature table
#define NH   (NN * HH)            // 160,000
#define EH   ((long long)EE * HH) // 5,120,000
#define MT   (NN / 16)            // 1250 row tiles

typedef __attribute__((ext_vector_type(16))) _Float16 v16h;
typedef __attribute__((ext_vector_type(8)))  float    v8f;

// ---------------------------------------------------------------------------
// WMMA helper (CDNA5 wave32, v_wmma_f32_16x16x32_f16)
// ---------------------------------------------------------------------------
__device__ __forceinline__ v8f wmma_f16(v16h a, v16h b, v8f c) {
    return __builtin_amdgcn_wmma_f32_16x16x32_f16(
        /*neg_a=*/false, a, /*neg_b=*/false, b,
        /*c_mod=*/(short)0, c, /*reuse_a=*/false, /*reuse_b=*/false);
}

// A-matrix 16x32 (MxK) f16 fragment from row-major fp32 tile (leading dim ld).
// lane<16: row M=lane, halves 0..7 -> K 0..7, halves 8..15 -> K 16..23
// lane>=16: row M=lane-16, K 8..15 and K 24..31          (ISA 7.12.2)
__device__ __forceinline__ v16h load_a_frag(const float* __restrict__ tile,
                                            int ld, int lane) {
    v16h a;
    const int m     = lane & 15;
    const int kbase = (lane & 16) ? 8 : 0;
    const float* rp = tile + m * ld;
#pragma unroll
    for (int i = 0; i < 16; ++i) {
        const int k = kbase + (i & 7) + ((i & 8) ? 16 : 0);
        a[i] = (_Float16)rp[k];
    }
    return a;
}

// B-matrix 32x16 (KxN) f16 fragment from row-major fp32 W (leading dim ld).
// lane<16: col N=lane, K 0..7 and K 16..23; lane>=16: K 8..15 and K 24..31
__device__ __forceinline__ v16h load_b_frag(const float* __restrict__ tile,
                                            int ld, int lane) {
    v16h b;
    const int n     = lane & 15;
    const int kbase = (lane & 16) ? 8 : 0;
#pragma unroll
    for (int i = 0; i < 16; ++i) {
        b[i] = (_Float16)tile[(kbase + (i & 7) + ((i & 8) ? 16 : 0)) * ld + n];
    }
    return b;
}

// Packed-fragment accessors: fragment f, lane l lives at (f*32 + l)*16 halves
// (32 B per lane -> one aligned v16h load in the GEMM hot loop).
__device__ __forceinline__ v16h load_packed(const _Float16* __restrict__ p,
                                            int frag, int lane) {
    return *(const v16h*)(p + ((size_t)frag * 32 + lane) * 16);
}
__device__ __forceinline__ void store_packed(_Float16* __restrict__ p,
                                             int frag, int lane, v16h v) {
    *(v16h*)(p + ((size_t)frag * 32 + lane) * 16) = v;
}

// ---------------------------------------------------------------------------
// Orderable-uint encoding of float for atomicMax-based segment max
// ---------------------------------------------------------------------------
__device__ __forceinline__ unsigned enc_ord(float f) {
    int i = __float_as_int(f);
    return (i >= 0) ? ((unsigned)i | 0x80000000u) : ~(unsigned)i;
}
__device__ __forceinline__ float dec_ord(unsigned u) {
    int i = (u & 0x80000000u) ? (int)(u & 0x7fffffffu) : ~(int)u;
    float f = __int_as_float(i);
    if (!(f > -1.0e37f)) f = 0.0f; // reference: non-finite max -> 0
    return f;
}

// ---------------------------------------------------------------------------
// K0: workspace init (segment max/sum + accumulators)
// ---------------------------------------------------------------------------
__global__ void k0_init(unsigned* __restrict__ rowmax, float* __restrict__ rowsum,
                        float* __restrict__ hlp, float* __restrict__ hbp) {
    int t = blockIdx.x * blockDim.x + threadIdx.x;
    if (t < NH) { rowmax[t] = 0x007FFFFFu; /* enc(-inf) */ rowsum[t] = 0.0f; }
    if (t < NF) { hlp[t] = 0.0f; hbp[t] = 0.0f; }
}

// ---------------------------------------------------------------------------
// Fragment pre-pack kernels (one wave per fragment; f32 -> f16 once)
// ---------------------------------------------------------------------------
// A-side pack: frag index = mt*kchunks + kc over a row-major [16*mtiles x ld]
__global__ void kp_pack_a(const float* __restrict__ src, int ld, int kchunks,
                          _Float16* __restrict__ dst) {
    const int wave = blockIdx.x * (blockDim.x >> 5) + (threadIdx.x >> 5);
    const int lane = threadIdx.x & 31;
    const int mt = wave / kchunks, kc = wave % kchunks;
    v16h a = load_a_frag(src + (size_t)mt * 16 * ld + kc * 32, ld, lane);
    store_packed(dst, wave, lane, a);
}

// B-side pack: frag index = kc*ntiles + nt over row-major W [32*kchunks x ld]
__global__ void kp_pack_b(const float* __restrict__ W, int ld, int ntiles,
                          _Float16* __restrict__ dst) {
    const int wave = blockIdx.x * (blockDim.x >> 5) + (threadIdx.x >> 5);
    const int lane = threadIdx.x & 31;
    const int kc = wave / ntiles, nt = wave % ntiles;
    v16h b = load_b_frag(W + (size_t)kc * 32 * ld + nt * 16, ld, lane);
    store_packed(dst, wave, lane, b);
}

// concat(h_lp, h_bp) A-side pack for the output GEMM: frag = mt*8 + kc
__global__ void kp_pack_h(const float* __restrict__ hlp,
                          const float* __restrict__ hbp,
                          _Float16* __restrict__ dst) {
    const int wave = blockIdx.x * (blockDim.x >> 5) + (threadIdx.x >> 5);
    const int lane = threadIdx.x & 31;
    const int mt = wave >> 3, kc = wave & 7;
    const float* src = (kc < 4)
        ? hlp + (size_t)mt * 16 * FIN + kc * 32
        : hbp + (size_t)mt * 16 * FIN + (kc - 4) * 32;
    v16h a = load_a_frag(src, FIN, lane);
    store_packed(dst, wave, lane, a);
}

// ---------------------------------------------------------------------------
// K1: fused QKV projection from packed fragments.
//     30000 waves = 1250 row-tiles x (3 matrices x 8 col-tiles).  K=128 -> 4 WMMA.
// ---------------------------------------------------------------------------
__global__ void k1_qkv_gemm(const _Float16* __restrict__ xpack,
                            const _Float16* __restrict__ wqp,
                            const _Float16* __restrict__ wkp,
                            const _Float16* __restrict__ wvp,
                            const float* __restrict__ bq,
                            const float* __restrict__ bk,
                            const float* __restrict__ bv,
                            float* __restrict__ qb, float* __restrict__ kb,
                            float* __restrict__ vb) {
    const int wave = blockIdx.x * (blockDim.x >> 5) + (threadIdx.x >> 5);
    const int lane = threadIdx.x & 31;
    const int mt   = wave / 24;        // 0..1249
    const int rem  = wave % 24;
    const int mat  = rem >> 3;         // 0=q 1=k 2=v
    const int nt   = rem & 7;

    const _Float16* wp   = (mat == 0) ? wqp : (mat == 1) ? wkp : wvp;
    const float*    bias = (mat == 0) ? bq  : (mat == 1) ? bk  : bv;
    float*          out  = (mat == 0) ? qb  : (mat == 1) ? kb  : vb;

    v8f c = {};
#pragma unroll
    for (int kc = 0; kc < 4; ++kc) {
        v16h a = load_packed(xpack, mt * 4 + kc, lane);
        v16h b = load_packed(wp,    kc * 8 + nt, lane);
        c = wmma_f16(a, b, c);
    }
    const int n     = nt * 16 + (lane & 15);
    const int mbase = mt * 16 + ((lane & 16) ? 8 : 0);
    const float bn  = bias[n];
#pragma unroll
    for (int r = 0; r < 8; ++r) {
        out[(size_t)(mbase + r) * FIN + n] = c[r] + bn;
    }
}

// ---------------------------------------------------------------------------
// K2: per-(edge,head) logits + segment max (orderable-uint atomicMax)
// ---------------------------------------------------------------------------
__global__ void k2_logits(const long long* __restrict__ ei,
                          const float* __restrict__ qb, const float* __restrict__ kb,
                          float* __restrict__ wbuf, unsigned* __restrict__ rowmax) {
    const int t = blockIdx.x * blockDim.x + threadIdx.x;
    const int e = t >> 3, h = t & 7;
    const int row = (int)ei[e];
    const int col = (int)ei[EE + e];
    const float4* qp = (const float4*)(qb + (size_t)row * FIN + h * DD);
    const float4* kp = (const float4*)(kb + (size_t)col * FIN + h * DD);
    float dot = 0.0f;
#pragma unroll
    for (int i = 0; i < 4; ++i) {
        float4 q4 = qp[i], k4 = kp[i];
        dot += q4.x * k4.x + q4.y * k4.y + q4.z * k4.z + q4.w * k4.w;
    }
    const float logit = dot * 0.25f; // 1/sqrt(16)
    wbuf[t] = logit;
    atomicMax(&rowmax[row * HH + h], enc_ord(logit));
}

// ---------------------------------------------------------------------------
// K3: exp(logit - max) + segment sum
// ---------------------------------------------------------------------------
__global__ void k3_expsum(const long long* __restrict__ ei,
                          const unsigned* __restrict__ rowmax,
                          float* __restrict__ wbuf, float* __restrict__ rowsum) {
    const int t = blockIdx.x * blockDim.x + threadIdx.x;
    const int e = t >> 3, h = t & 7;
    const int row = (int)ei[e];
    const float m = dec_ord(rowmax[row * HH + h]);
    const float w = __expf(wbuf[t] - m);
    wbuf[t] = w;
    atomicAdd(&rowsum[row * HH + h], w);
}

// ---------------------------------------------------------------------------
// K4: edge aggregation: h_lp += alpha*v_j ; h_bp += alpha*tanh(v_j@Wpsi+bpsi)
// ---------------------------------------------------------------------------
__global__ void k4_aggregate(const long long* __restrict__ ei,
                             const float* __restrict__ vb,
                             const float* __restrict__ wbuf,
                             const float* __restrict__ rowsum,
                             const float* __restrict__ Wpsi,
                             const float* __restrict__ bpsi,
                             float* __restrict__ hlp, float* __restrict__ hbp) {
    __shared__ float sW[DD * DD];
    __shared__ float sb[DD];
    if (threadIdx.x < DD * DD) sW[threadIdx.x] = Wpsi[threadIdx.x];
    if (threadIdx.x < DD)      sb[threadIdx.x] = bpsi[threadIdx.x];
    __syncthreads();

    const int t = blockIdx.x * blockDim.x + threadIdx.x;
    const int e = t >> 3, h = t & 7;
    const int row = (int)ei[e];
    const int col = (int)ei[EE + e];
    const float alpha = wbuf[t] / (rowsum[row * HH + h] + 1e-16f);

    float vj[DD];
    const float4* vp = (const float4*)(vb + (size_t)col * FIN + h * DD);
#pragma unroll
    for (int i = 0; i < 4; ++i) {
        float4 v4 = vp[i];
        vj[4 * i + 0] = v4.x; vj[4 * i + 1] = v4.y;
        vj[4 * i + 2] = v4.z; vj[4 * i + 3] = v4.w;
    }
    const int base = row * FIN + h * DD;
#pragma unroll
    for (int j = 0; j < DD; ++j) {
        float acc = sb[j];
#pragma unroll
        for (int i = 0; i < DD; ++i) acc += vj[i] * sW[i * DD + j];
        const float vbpj = tanhf(acc);
        atomicAdd(&hlp[base + j], alpha * vj[j]);
        atomicAdd(&hbp[base + j], alpha * vbpj);
    }
}

// ---------------------------------------------------------------------------
// K5: output projection from packed fragments (K=256 -> 8 WMMA) fused with
//     bias + ReLU + residual.  10000 waves.
// ---------------------------------------------------------------------------
__global__ void k5_out_gemm(const _Float16* __restrict__ hpack,
                            const _Float16* __restrict__ wop,
                            const float* __restrict__ bo,
                            const float* __restrict__ x, float* __restrict__ y) {
    const int wave = blockIdx.x * (blockDim.x >> 5) + (threadIdx.x >> 5);
    const int lane = threadIdx.x & 31;
    const int mt   = wave >> 3;   // 0..1249
    const int nt   = wave & 7;

    v8f c = {};
#pragma unroll
    for (int kc = 0; kc < 8; ++kc) {
        v16h a = load_packed(hpack, mt * 8 + kc, lane);
        v16h b = load_packed(wop,   kc * 8 + nt, lane);
        c = wmma_f16(a, b, c);
    }
    const int n     = nt * 16 + (lane & 15);
    const int mbase = mt * 16 + ((lane & 16) ? 8 : 0);
    const float bn  = bo[n];
#pragma unroll
    for (int r = 0; r < 8; ++r) {
        const size_t idx = (size_t)(mbase + r) * FIN + n;
        const float hout = c[r] + bn;
        y[idx] = x[idx] + fmaxf(hout, 0.0f);
    }
}

// ---------------------------------------------------------------------------
// K6: per-row LayerNorm (128 threads per row)
// ---------------------------------------------------------------------------
__global__ void k6_layernorm(const float* __restrict__ y,
                             const float* __restrict__ gamma,
                             const float* __restrict__ beta,
                             float* __restrict__ out) {
    __shared__ float red[FIN];
    const int r = blockIdx.x;
    const int f = threadIdx.x;
    const float v = y[(size_t)r * FIN + f];

    red[f] = v;
    __syncthreads();
#pragma unroll
    for (int s = FIN / 2; s > 0; s >>= 1) {
        if (f < s) red[f] += red[f + s];
        __syncthreads();
    }
    const float mu = red[0] * (1.0f / FIN);
    __syncthreads();

    const float d = v - mu;
    red[f] = d * d;
    __syncthreads();
#pragma unroll
    for (int s = FIN / 2; s > 0; s >>= 1) {
        if (f < s) red[f] += red[f + s];
        __syncthreads();
    }
    const float var = red[0] * (1.0f / FIN);
    out[(size_t)r * FIN + f] = d * rsqrtf(var + 1e-5f) * gamma[f] + beta[f];
}

// ---------------------------------------------------------------------------
// Host launcher
// ---------------------------------------------------------------------------
extern "C" void kernel_launch(void* const* d_in, const int* in_sizes, int n_in,
                              void* d_out, int out_size, void* d_ws, size_t ws_size,
                              hipStream_t stream) {
    // Inputs in setup_inputs() order
    const float*     x    = (const float*)d_in[0];
    const long long* ei   = (const long long*)d_in[1];  // int64 edge_index [2,E]
    const float*     Wq   = (const float*)d_in[2];
    const float*     bq   = (const float*)d_in[3];
    const float*     Wk   = (const float*)d_in[4];
    const float*     bk   = (const float*)d_in[5];
    const float*     Wv   = (const float*)d_in[6];
    const float*     bv   = (const float*)d_in[7];
    const float*     Wpsi = (const float*)d_in[8];
    const float*     bpsi = (const float*)d_in[9];
    const float*     Wo   = (const float*)d_in[10];
    const float*     bo   = (const float*)d_in[11];
    const float*     gam  = (const float*)d_in[12];
    const float*     bet  = (const float*)d_in[13];
    float*           out  = (float*)d_out;

    // Workspace layout: float region first, then 32B-aligned f16 pack region
    float* ws = (float*)d_ws;
    float*    qb     = ws;                      // NF
    float*    kb     = qb + NF;                 // NF
    float*    vb     = kb + NF;                 // NF
    float*    wbuf   = vb + NF;                 // E*H
    unsigned* rowmax = (unsigned*)(wbuf + EH);  // N*H
    float*    rowsum = (float*)(rowmax + NH);   // N*H
    float*    hlp    = rowsum + NH;             // NF
    float*    hbp    = hlp + NF;                // NF
    float*    ybuf   = hbp + NF;                // NF
    _Float16* xpack  = (_Float16*)(ybuf + NF);  // 1250*4*512   halves (32B aligned)
    _Float16* wqp    = xpack + (size_t)MT * 4 * 512; // 4*8*512
    _Float16* wkp    = wqp + 4 * 8 * 512;
    _Float16* wvp    = wkp + 4 * 8 * 512;
    _Float16* wop    = wvp + 4 * 8 * 512;            // 8*8*512
    _Float16* hpack  = wop + 8 * 8 * 512;            // 1250*8*512
    (void)ws_size; (void)in_sizes; (void)n_in; (void)out_size;

    const int TPB = 256; // 8 waves per block

    // K0: init reductions/accumulators
    k0_init<<<(NF + TPB - 1) / TPB, TPB, 0, stream>>>(rowmax, rowsum, hlp, hbp);

    // One-shot f32->f16 fragment packing (x + all weight matrices)
    kp_pack_a<<<MT * 4 / 8, TPB, 0, stream>>>(x, FIN, 4, xpack);      // 5000 waves
    kp_pack_b<<<32 / 8, TPB, 0, stream>>>(Wq, FIN, 8, wqp);           // 32 waves
    kp_pack_b<<<32 / 8, TPB, 0, stream>>>(Wk, FIN, 8, wkp);
    kp_pack_b<<<32 / 8, TPB, 0, stream>>>(Wv, FIN, 8, wvp);
    kp_pack_b<<<64 / 8, TPB, 0, stream>>>(Wo, FIN, 8, wop);           // 64 waves

    // K1: QKV projection (30000 waves)
    k1_qkv_gemm<<<30000 / 8, TPB, 0, stream>>>(xpack, wqp, wkp, wvp,
                                               bq, bk, bv, qb, kb, vb);

    // K2: edge logits + segment max
    k2_logits<<<(int)(EH / TPB), TPB, 0, stream>>>(ei, qb, kb, wbuf, rowmax);

    // K3: exp + segment sum
    k3_expsum<<<(int)(EH / TPB), TPB, 0, stream>>>(ei, rowmax, wbuf, rowsum);

    // K4: edge aggregation (alpha*v, alpha*tanh(v@Wpsi+bpsi))
    k4_aggregate<<<(int)(EH / TPB), TPB, 0, stream>>>(ei, vb, wbuf, rowsum,
                                                      Wpsi, bpsi, hlp, hbp);

    // Pack concat(h_lp,h_bp) A-fragments for the output GEMM (10000 waves)
    kp_pack_h<<<MT * 8 / 8, TPB, 0, stream>>>(hlp, hbp, hpack);

    // K5: output GEMM + bias + ReLU + residual (10000 waves)
    k5_out_gemm<<<10000 / 8, TPB, 0, stream>>>(hpack, wop, bo, x, ybuf);

    // K6: LayerNorm -> d_out
    k6_layernorm<<<NN, FIN, 0, stream>>>(ybuf, gam, bet, out);
}
